// MoCo_86397562126951
// MI455X (gfx1250) — matrix operand order repaired and checked by reading, not compiled, not run on hardware
//
#include <hip/hip_runtime.h>
#include <hip/hip_bf16.h>
#include <math.h>

// ---- problem constants (match reference) ----
#define B2  512
#define NN  49
#define DD  256
#define CK  128   // C and K both 128
#define TAU_T    0.04f
#define TAU_S    0.1f
#define MPAD 64   // 49 padded to 64 for 16x16 WMMA tiling

typedef __attribute__((ext_vector_type(16))) _Float16 v16h;
typedef __attribute__((ext_vector_type(8)))  _Float16 v8h;
typedef __attribute__((ext_vector_type(8)))  float    v8f;

__device__ __forceinline__ float wsum(float v) {
#pragma unroll
  for (int o = 16; o > 0; o >>= 1) v += __shfl_xor(v, o, 32);
  return v;
}
__device__ __forceinline__ float wmaxr(float v) {
#pragma unroll
  for (int o = 16; o > 0; o >>= 1) v = fmaxf(v, __shfl_xor(v, o, 32));
  return v;
}

// ---------------------------------------------------------------------------
// Kernel 1: dequeue — gather/select tc and tr straight into the output buffer.
// ---------------------------------------------------------------------------
__global__ void __launch_bounds__(256)
dequeue_kernel(const float* __restrict__ t_cls, const float* __restrict__ t_reg,
               const float* __restrict__ q_all, const float* __restrict__ q_grid,
               const int* __restrict__ cidx, const int* __restrict__ pidx,
               const int* __restrict__ useq,
               float* __restrict__ out_tc, float* __restrict__ out_tr) {
  const int b = blockIdx.x;
  const int t = threadIdx.x;
  const bool uq = (useq[b] != 0);
  const size_t q = (size_t)cidx[b] * CK + (size_t)pidx[b];

  // tc: 256 floats, one per thread
  {
    const float* src = uq ? (q_all + q * DD) : (t_cls + (size_t)b * DD);
    out_tc[(size_t)b * DD + t] = src[t];
  }
  // tr: 49*256 floats, float4 vectorized
  {
    const float4* src = uq ? (const float4*)(q_grid + q * (size_t)(2 * NN) * DD)
                           : (const float4*)(t_reg + (size_t)b * NN * DD);
    float4* dst = (float4*)(out_tr + (size_t)b * NN * DD);
    for (int i = t; i < NN * DD / 4; i += 256) dst[i] = src[i];
  }
}

// ---------------------------------------------------------------------------
// Kernel 2: per batch-row loss. One block (8 wave32s) per teacher row r,
// paired with student row p = (r+256)%512.
// ---------------------------------------------------------------------------
__global__ void __launch_bounds__(256)
esvit_kernel(const float* __restrict__ s_cls, const float* __restrict__ s_reg_raw,
             const float* __restrict__ s_fea, const float* __restrict__ t_fea,
             const float* __restrict__ q_grid,
             const float* __restrict__ center, const float* __restrict__ cgrid,
             const int* __restrict__ cidx, const int* __restrict__ pidx,
             const int* __restrict__ useq,
             const float* __restrict__ tc, const float* __restrict__ tr,
             float* __restrict__ partial) {
  extern __shared__ char smem[];
  _Float16* sfh = (_Float16*)smem;                               // 64*256 f16 (32KB)
  _Float16* tfh = (_Float16*)(smem + MPAD * DD * 2);             // 64*256 f16 (32KB)
  float*    sim = (float*)(smem + 2 * MPAD * DD * 2);            // 64*64 f32  (16KB)
  int*      ind = (int*)  (smem + 2 * MPAD * DD * 2 + MPAD * MPAD * 4);
  float*    red = (float*)(smem + 2 * MPAD * DD * 2 + MPAD * MPAD * 4 + MPAD * 4);

  const int r = blockIdx.x;
  const int p = (r + B2 / 2) & (B2 - 1);
  const int w = threadIdx.x >> 5;
  const int lane = threadIdx.x & 31;
  const bool uq = (useq[r] != 0);
  const size_t qoff = ((size_t)cidx[r] * CK + (size_t)pidx[r]) * (size_t)(2 * NN) * DD;

  // ---- Phase 1: l2-normalize fea rows -> f16 in LDS (wave per row) ----
  for (int row = w; row < MPAD; row += 8) {
    if (row < NN) {
      const float* sp = s_fea + ((size_t)p * NN + row) * DD + lane * 8;
      const float* tp = uq ? (q_grid + qoff + (size_t)(NN + row) * DD + lane * 8)
                           : (t_fea + ((size_t)r * NN + row) * DD + lane * 8);
      float sv[8], tv[8], ss = 0.f, ts = 0.f;
#pragma unroll
      for (int j = 0; j < 8; j++) {
        sv[j] = sp[j]; ss += sv[j] * sv[j];
        tv[j] = tp[j]; ts += tv[j] * tv[j];
      }
      ss = wsum(ss); ts = wsum(ts);
      const float si = 1.0f / fmaxf(sqrtf(ss), 1e-12f);
      const float ti = 1.0f / fmaxf(sqrtf(ts), 1e-12f);
#pragma unroll
      for (int j = 0; j < 8; j++) {
        sfh[row * DD + lane * 8 + j] = (_Float16)(sv[j] * si);
        tfh[row * DD + lane * 8 + j] = (_Float16)(tv[j] * ti);
      }
    } else {
#pragma unroll
      for (int j = 0; j < 8; j++) {
        sfh[row * DD + lane * 8 + j] = (_Float16)0.f;
        tfh[row * DD + lane * 8 + j] = (_Float16)0.f;
      }
    }
  }
  __syncthreads();

  // ---- Phase 2: sim = Snorm x Tnorm^T via v_wmma_f32_16x16x32_f16 ----
  // A (16x32 f16): lane<16 holds K {0..7,16..23}; lane>=16 holds {8..15,24..31}
  // B (32x16 f16): lane<16 holds K {0..7,16..23}; lane>=16 holds {8..15,24..31}; col = lane%16
  const int klA = (lane >= 16) ? 8 : 0;
  const int klB = (lane >= 16) ? 8 : 0;
#pragma unroll
  for (int tt = 0; tt < 2; ++tt) {
    const int tile = w + tt * 8;          // 16 tiles over 8 waves
    const int mt = tile >> 2, nt = tile & 3;
    const int arow = mt * 16 + (lane & 15);
    const int bcol = nt * 16 + (lane & 15);
    v8f acc = {};
#pragma unroll
    for (int kk = 0; kk < DD; kk += 32) {
      const v8h alo = *(const v8h*)(sfh + arow * DD + kk + klA);
      const v8h ahi = *(const v8h*)(sfh + arow * DD + kk + klA + 16);
      const v8h blo = *(const v8h*)(tfh + bcol * DD + kk + klB);
      const v8h bhi = *(const v8h*)(tfh + bcol * DD + kk + klB + 16);
      v16h a, bm;
#pragma unroll
      for (int j = 0; j < 8; j++) {
        a[j] = alo[j]; a[j + 8] = ahi[j];
        bm[j] = blo[j]; bm[j + 8] = bhi[j];
      }
      acc = __builtin_amdgcn_wmma_f32_16x16x32_f16(false, a, false, bm,
                                                   (short)0, acc, false, false);
    }
    // C layout: M = v + 8*(lane>=16), N = lane%16
    const int mbase = mt * 16 + ((lane >= 16) ? 8 : 0);
    const int ncol  = nt * 16 + (lane & 15);
#pragma unroll
    for (int v = 0; v < 8; ++v) sim[(mbase + v) * MPAD + ncol] = acc[v];
  }
  __syncthreads();

  // ---- Phase 3: argmax over teacher tokens (first-max, matches jnp.argmax) ----
  if (threadIdx.x < NN) {
    const int n = threadIdx.x;
    float best = sim[n * MPAD]; int bi = 0;
    for (int m = 1; m < NN; m++) {
      const float v = sim[n * MPAD + m];
      if (v > best) { best = v; bi = m; }
    }
    ind[n] = bi;
  }
  __syncthreads();

  // ---- Phase 4: fused teacher-softmax / student-log-softmax CE ----
  float wacc = 0.f;
  for (int task = w; task < NN + 1; task += 8) {   // 49 grid rows + 1 cls row
    const float *trow, *srow, *cen;
    float sinv, wgt;
    if (task < NN) {
      trow = tr + ((size_t)r * NN + ind[task]) * DD;
      srow = s_reg_raw + ((size_t)p * NN + task) * DD;
      cen = cgrid; sinv = 1.0f / TAU_S; wgt = 0.5f / (float)NN;
    } else {
      trow = tc + (size_t)r * DD;
      srow = s_cls + (size_t)p * DD;
      cen = center; sinv = 1.0f / TAU_T; wgt = 0.5f;
    }
    float tl[8], sl[8], tm = -3.402823466e38f, sm = -3.402823466e38f;
#pragma unroll
    for (int j = 0; j < 8; j++) {
      const int d = lane * 8 + j;
      tl[j] = (trow[d] - cen[d]) * (1.0f / TAU_T);
      sl[j] = srow[d] * sinv;
      tm = fmaxf(tm, tl[j]); sm = fmaxf(sm, sl[j]);
    }
    tm = wmaxr(tm); sm = wmaxr(sm);
    float tsum = 0.f, ssum = 0.f, tdot = 0.f;
#pragma unroll
    for (int j = 0; j < 8; j++) {
      const float e = __expf(tl[j] - tm);
      tsum += e;
      ssum += __expf(sl[j] - sm);
      tdot += e * sl[j];
    }
    tsum = wsum(tsum); ssum = wsum(ssum); tdot = wsum(tdot);
    // CE = -sum(pT * logsoftmax(S)) = sm + log(ssum) - sum(pT*S)
    const float ce = sm + __logf(ssum) - tdot / tsum;
    if (lane == 0) wacc += wgt * ce;
  }
  if (lane == 0) red[w] = wacc;
  __syncthreads();
  if (threadIdx.x == 0) {
    float s = 0.f;
#pragma unroll
    for (int i = 0; i < 8; i++) s += red[i];
    partial[r] = s;
  }
}

// ---------------------------------------------------------------------------
// Kernel 3: deterministic fixed-order reduction of 512 partials -> loss.
// ---------------------------------------------------------------------------
__global__ void __launch_bounds__(256)
reduce_kernel(const float* __restrict__ partial, float* __restrict__ out) {
  __shared__ float sh[256];
  const int t = threadIdx.x;
  sh[t] = partial[t] + partial[t + 256];
  __syncthreads();
  for (int s = 128; s > 0; s >>= 1) {
    if (t < s) sh[t] += sh[t + s];
    __syncthreads();
  }
  if (t == 0) out[0] = sh[0] * (1.0f / (float)B2);
}

extern "C" void kernel_launch(void* const* d_in, const int* in_sizes, int n_in,
                              void* d_out, int out_size, void* d_ws, size_t ws_size,
                              hipStream_t stream) {
  (void)in_sizes; (void)n_in; (void)out_size; (void)ws_size;
  const float* s_cls  = (const float*)d_in[0];
  const float* s_reg  = (const float*)d_in[1];
  const float* s_fea  = (const float*)d_in[2];
  const float* t_cls  = (const float*)d_in[3];
  const float* t_reg  = (const float*)d_in[4];
  const float* t_fea  = (const float*)d_in[5];
  const float* q_all  = (const float*)d_in[6];
  const float* q_grid = (const float*)d_in[7];
  const float* center = (const float*)d_in[8];
  const float* cgrid  = (const float*)d_in[9];
  const int*   cidx   = (const int*)d_in[10];
  const int*   pidx   = (const int*)d_in[11];
  const int*   useq   = (const int*)d_in[12];

  float* out_loss = (float*)d_out;                 // [1]
  float* out_tc   = out_loss + 1;                  // [512*256]
  float* out_tr   = out_tc + (size_t)B2 * DD;      // [512*49*256]
  float* partial  = (float*)d_ws;                  // [512] scratch

  dequeue_kernel<<<B2, 256, 0, stream>>>(t_cls, t_reg, q_all, q_grid,
                                         cidx, pidx, useq, out_tc, out_tr);

  const size_t lds = 2 * (size_t)MPAD * DD * sizeof(_Float16)   // sfh + tfh
                   + (size_t)MPAD * MPAD * sizeof(float)        // sim
                   + MPAD * sizeof(int)                         // ind
                   + 8 * sizeof(float);                         // red
  esvit_kernel<<<B2, 256, lds, stream>>>(s_cls, s_reg, s_fea, t_fea, q_grid,
                                         center, cgrid, cidx, pidx, useq,
                                         out_tc, out_tr, partial);

  reduce_kernel<<<1, 256, 0, stream>>>(partial, out_loss);
}
